// VQVAE_80032420594181
// MI455X (gfx1250) — matrix operand order repaired and checked by reading, compiled, not run on hardware
//
#include <hip/hip_runtime.h>

// ---------------------------------------------------------------------------
// VQ-VAE codebook lookup for MI455X (gfx1250, wave32, WMMA + TDM).
//   scores S = X @ E  via split-bf16 WMMA (xh*eh + xl*eh + xh*el ~ fp32)
//   idx = argmax_k (2*S - ||e_k||^2);  out = E^T[idx] gathered in exact fp32.
// Codebook tiles are staged into LDS by the Tensor Data Mover (one
// tensor_load_to_lds per table, hardware row padding), tracked by TENSORcnt.
// ---------------------------------------------------------------------------

typedef __attribute__((ext_vector_type(16))) __bf16    v16bf;
typedef __attribute__((ext_vector_type(8)))  float     v8f;
typedef __attribute__((ext_vector_type(4)))  unsigned  uint32x4;
typedef __attribute__((ext_vector_type(8)))  unsigned  uint32x8;

#define N_TOTAL   (32 * 64 * 64)   // 131072 vectors
#define D_DIM     128
#define K_CODES   1024
#define KC        256              // codes staged in LDS per chunk
#define ROW_STRIDE 272             // bytes per code row in LDS (256B + 16B pad -> conflict-free b128)
#define LDS_LO_BASE (KC * ROW_STRIDE)
#define LDS_BYTES   (2 * KC * ROW_STRIDE)   // 139264 B (dynamic LDS; 320KB/WGP on CDNA5)

// ---- workspace layout (d_ws) ----
#define WS_EFT_OFF    0                      // fp32 E^T [K][D]   512 KB
#define WS_EHT_OFF    (512 * 1024)           // bf16 hi E^T       256 KB
#define WS_ELT_OFF    (768 * 1024)           // bf16 lo E^T       256 KB
#define WS_ENORM_OFF  (1024 * 1024)          // fp32 ||e||^2        4 KB
#define WS_IDX_OFF    (1024 * 1024 + 4096)   // int32 idx[N]      512 KB

// ---------------------------------------------------------------------------
// Kernel 1: transpose codebook, split to bf16 hi/lo, compute ||e||^2.
// ---------------------------------------------------------------------------
__global__ void vq_prep(const float* __restrict__ E, float* __restrict__ EfT,
                        __bf16* __restrict__ EhT, __bf16* __restrict__ ElT,
                        float* __restrict__ eNorm) {
  const int k = blockIdx.x * blockDim.x + threadIdx.x;
  if (k >= K_CODES) return;
  float nrm = 0.0f;
  for (int d = 0; d < D_DIM; ++d) {
    const float v = E[(size_t)d * K_CODES + k];   // coalesced across k
    nrm += v * v;
    EfT[k * D_DIM + d] = v;
    const __bf16 hi = (__bf16)v;
    EhT[k * D_DIM + d] = hi;
    ElT[k * D_DIM + d] = (__bf16)(v - (float)hi);
  }
  eNorm[k] = nrm;
}

// ---------------------------------------------------------------------------
// TDM: DMA a [KC x 128] bf16 2D tile (contiguous rows of 256B) into LDS with
// a 16B pad inserted after every 256B (pad_interval=5 -> 8*2^5=256B,
// pad_amount=3 -> 4 DWORDs), i.e. LDS row stride = ROW_STRIDE.
// D# group0: count=1, lds_addr, global_addr[56:0], type=2.
// D# group1: data_size=1(2B), pad_enable, dims/strides for a 2D tensor.
// ---------------------------------------------------------------------------
__device__ __forceinline__ void tdm_load_tile(const void* gsrc, unsigned lds_off) {
  const unsigned long long ga = (unsigned long long)(size_t)gsrc;
  union { unsigned u[4]; uint32x4 v; } g0;
  union { unsigned u[8]; uint32x8 v; } g1;
  g0.u[0] = 1u;                                              // count=1, user mode
  g0.u[1] = lds_off;                                         // lds_addr [63:32]
  g0.u[2] = (unsigned)ga;                                    // global_addr [95:64]
  g0.u[3] = (unsigned)((ga >> 32) & 0x01FFFFFFull)           // global_addr [120:96]
          | (2u << 30);                                      // type=2 ("image")
  g1.u[0] = (1u << 16)                                       // data_size = 2 bytes
          | (1u << 20)                                       // pad_enable
          | (5u << 22)                                       // pad_interval: 256B
          | (3u << 25);                                      // pad_amount: 4 DWORDs
  g1.u[1] = ((unsigned)D_DIM & 0xFFFFu) << 16;               // tensor_dim0[15:0]
  g1.u[2] = ((unsigned)KC & 0xFFFFu) << 16;                  // dim0 hi=0 | tensor_dim1[15:0]
  g1.u[3] = ((unsigned)D_DIM) << 16;                         // dim1 hi=0 | tile_dim0=128
  g1.u[4] = (unsigned)KC;                                    // tile_dim1=KC | tile_dim2=0
  g1.u[5] = (unsigned)D_DIM;                                 // tensor_dim0_stride lo32
  g1.u[6] = (((unsigned)(D_DIM * KC)) & 0xFFFFu) << 16;      // d0s hi=0 | d1s[15:0]
  g1.u[7] = (unsigned)((D_DIM * KC) >> 16);                  // tensor_dim1_stride[47:16]
  asm volatile("tensor_load_to_lds %0, %1" :: "s"(g0.v), "s"(g1.v) : "memory");
}

// ---------------------------------------------------------------------------
// Kernel 2: WMMA score GEMM + running argmax.
// Block = 256 threads (8 wave32s). Wave w owns rows [blk*128 + 16w, +16).
// ---------------------------------------------------------------------------
__global__ __launch_bounds__(256)
void vq_argmax(const float* __restrict__ X, const __bf16* __restrict__ EhT,
               const __bf16* __restrict__ ElT, const float* __restrict__ eNorm,
               int* __restrict__ idxOut) {
  extern __shared__ unsigned char lds[];
  const int t     = threadIdx.x;
  const int wave  = t >> 5;
  const int lane  = t & 31;
  const int laneN = lane & 15;     // column slot within a 16-wide tile
  const int half  = lane >> 4;     // 0: lanes 0-15, 1: lanes 16-31
  const int rbase = blockIdx.x * 128 + wave * 16;
  const int row   = rbase + laneN; // A-matrix: lane L and L+16 both feed row (L&15)

  // ---- Build A fragments (16 rows x 128 deep) in registers, hi/lo bf16.
  // ISA 16-bit A 16x32 layout: lane holds K = {8h..8h+7, 16+8h..16+8h+7}.
  v16bf Ah[4], Al[4];
  const float4* xr = (const float4*)(X + (size_t)row * D_DIM);
  #pragma unroll
  for (int c = 0; c < 4; ++c) {
    float tmp[16];
    const float4 p0 = xr[8 * c + 2 * half];
    const float4 p1 = xr[8 * c + 2 * half + 1];
    const float4 p2 = xr[8 * c + 4 + 2 * half];
    const float4 p3 = xr[8 * c + 4 + 2 * half + 1];
    tmp[0] = p0.x;  tmp[1] = p0.y;  tmp[2]  = p0.z;  tmp[3]  = p0.w;
    tmp[4] = p1.x;  tmp[5] = p1.y;  tmp[6]  = p1.z;  tmp[7]  = p1.w;
    tmp[8] = p2.x;  tmp[9] = p2.y;  tmp[10] = p2.z;  tmp[11] = p2.w;
    tmp[12] = p3.x; tmp[13] = p3.y; tmp[14] = p3.z;  tmp[15] = p3.w;
    #pragma unroll
    for (int j = 0; j < 16; ++j) {
      const __bf16 hi = (__bf16)tmp[j];
      Ah[c][j] = hi;
      Al[c][j] = (__bf16)(tmp[j] - (float)hi);
    }
  }

  float best[8];
  int   bidx[8];
  #pragma unroll
  for (int v = 0; v < 8; ++v) { best[v] = -3.4e38f; bidx[v] = 0; }

  for (int kc = 0; kc < K_CODES / KC; ++kc) {
    __syncthreads();   // previous chunk fully consumed by all waves
    if (wave == 0) {   // one wave drives the DMA engine for the block
      tdm_load_tile(EhT + (size_t)kc * KC * D_DIM, 0u);
      tdm_load_tile(ElT + (size_t)kc * KC * D_DIM, (unsigned)LDS_LO_BASE);
      __builtin_amdgcn_s_wait_tensorcnt(0);
    }
    __syncthreads();   // publish DMA'd tile to all waves

    for (int kt = 0; kt < KC / 16; ++kt) {
      const int codeLocal = kt * 16 + laneN;
      // ISA 16-bit B 32x16 layout: lane holds K = {8h..8h+7, 16+8h..16+8h+7} of col laneN.
      const unsigned bbase = (unsigned)codeLocal * ROW_STRIDE + (unsigned)half * 16;

      // Preload all 8 B fragments so WMMAs can issue back-to-back.
      v16bf Bh[4], Bl[4];
      #pragma unroll
      for (int c = 0; c < 4; ++c) {
        union { uint4 q[2]; v16bf v; } ubh, ubl;
        const unsigned off = bbase + 64u * c;
        ubh.q[0] = *(const uint4*)(lds + off);
        ubh.q[1] = *(const uint4*)(lds + off + 32);
        ubl.q[0] = *(const uint4*)(lds + LDS_LO_BASE + off);
        ubl.q[1] = *(const uint4*)(lds + LDS_LO_BASE + off + 32);
        Bh[c] = ubh.v;
        Bl[c] = ubl.v;
      }

      // Two accumulator chains for matrix-pipe ILP; split-bf16:
      // xh*eh + xl*eh + xh*el (xl*el term negligible).
      v8f acc0 = {}, acc1 = {};
      #pragma unroll
      for (int c = 0; c < 2; ++c) {
        acc0 = __builtin_amdgcn_wmma_f32_16x16x32_bf16(false, Ah[c], false, Bh[c],
                                                       (short)0, acc0, false, false);
        acc1 = __builtin_amdgcn_wmma_f32_16x16x32_bf16(false, Ah[c + 2], false, Bh[c + 2],
                                                       (short)0, acc1, false, false);
        acc0 = __builtin_amdgcn_wmma_f32_16x16x32_bf16(false, Al[c], false, Bh[c],
                                                       (short)0, acc0, false, false);
        acc1 = __builtin_amdgcn_wmma_f32_16x16x32_bf16(false, Al[c + 2], false, Bh[c + 2],
                                                       (short)0, acc1, false, false);
        acc0 = __builtin_amdgcn_wmma_f32_16x16x32_bf16(false, Ah[c], false, Bl[c],
                                                       (short)0, acc0, false, false);
        acc1 = __builtin_amdgcn_wmma_f32_16x16x32_bf16(false, Ah[c + 2], false, Bl[c + 2],
                                                       (short)0, acc1, false, false);
      }

      const int   col = kc * KC + codeLocal;
      const float en  = eNorm[col];
      // D layout: acc[v] is (row M = v + 8*half, col N = laneN) of this tile.
      #pragma unroll
      for (int v = 0; v < 8; ++v) {
        const float s = 2.0f * (acc0[v] + acc1[v]) - en;
        if (s > best[v]) { best[v] = s; bidx[v] = col; }
      }
    }
  }

  // ---- cross-lane argmax within each 16-lane half (columns of one row)
  #pragma unroll
  for (int v = 0; v < 8; ++v) {
    float s = best[v];
    int   bi = bidx[v];
    #pragma unroll
    for (int m = 1; m <= 8; m <<= 1) {
      const float s2 = __shfl_xor(s, m, 32);
      const int   i2 = __shfl_xor(bi, m, 32);
      if (s2 > s || (s2 == s && i2 < bi)) { s = s2; bi = i2; }
    }
    best[v] = s;
    bidx[v] = bi;
  }
  if (laneN == 0) {
    #pragma unroll
    for (int v = 0; v < 8; ++v) idxOut[rbase + v + 8 * half] = bidx[v];
  }
}

// ---------------------------------------------------------------------------
// Kernel 3: gather exact fp32 code vectors. 8 rows per 256-thread block,
// 32 lanes x float4 per row -> fully coalesced 512B row copies.
// ---------------------------------------------------------------------------
__global__ void vq_gather(const int* __restrict__ idx, const float* __restrict__ EfT,
                          float* __restrict__ out) {
  const int t   = threadIdx.x;
  const int n   = blockIdx.x * 8 + (t >> 5);
  const int e4  = t & 31;                       // 128 floats = 32 float4
  const int k   = idx[n];
  ((float4*)out)[(size_t)n * 32 + e4] = ((const float4*)EfT)[(size_t)k * 32 + e4];
}

// ---------------------------------------------------------------------------
extern "C" void kernel_launch(void* const* d_in, const int* in_sizes, int n_in,
                              void* d_out, int out_size, void* d_ws, size_t ws_size,
                              hipStream_t stream) {
  const float* X = (const float*)d_in[0];   // [B,H,W,D] fp32
  const float* E = (const float*)d_in[1];   // [D,K] fp32
  float* out = (float*)d_out;

  unsigned char* ws = (unsigned char*)d_ws;
  float*  EfT   = (float*)(ws + WS_EFT_OFF);
  __bf16* EhT   = (__bf16*)(ws + WS_EHT_OFF);
  __bf16* ElT   = (__bf16*)(ws + WS_ELT_OFF);
  float*  eNorm = (float*)(ws + WS_ENORM_OFF);
  int*    idx   = (int*)(ws + WS_IDX_OFF);

  vq_prep<<<K_CODES / 256, 256, 0, stream>>>(E, EfT, EhT, ElT, eNorm);
  vq_argmax<<<N_TOTAL / 128, 256, LDS_BYTES, stream>>>(X, EhT, ElT, eNorm, idx);
  vq_gather<<<N_TOTAL / 8, 256, 0, stream>>>(idx, EfT, out);
}
